// RNNRecurrentEntitiyDecoder_25494925869202
// MI455X (gfx1250) — matrix operand order repaired and checked by reading, not compiled, not run on hardware
//
#include <hip/hip_runtime.h>
#include <hip/hip_bf16.h>

// CDNA5 / gfx1250 wave32 WMMA recurrent entity decoder.
// One block per batch b. h (20x256 padded to 32x256) resides in LDS (f32 truth
// copy + bf16 shadow in A-fragment-friendly layout) for all 512 steps.
// U is register-resident as bf16 WMMA B-fragments (loaded once, reused 512x).
// x@W also runs on WMMA using W pre-packed in B-fragment order in LDS, with a
// zero-padded fragment staging buffer for x (zeros written once at setup).
// kV is pre-packed in C-fragment per-lane-contiguous layout for b128 reads.
// Steps with mask[b,t]==0 are skipped entirely (h unchanged per reference).

typedef __attribute__((ext_vector_type(16))) __bf16 v16bf;
typedef __attribute__((ext_vector_type(8)))  float  v8f;

#define BB   256
#define SS   512
#define KK   20
#define DD   256
#define HSTR 260          // f32 row stride (words): bank spread + 16B alignment
#define HB   264          // bf16 shadow row stride: 528B rows, 16B aligned
#define EPSN 1e-12f

// ---- dynamic LDS layout ----------------------------------------------------
constexpr int SWF_OFF  = 0;                         // bf16 W frag-packed     131072 B
constexpr int SH_OFF   = SWF_OFF + DD * DD * 2;     // f32 h[32][HSTR]         33280 B
constexpr int SHB_OFF  = SH_OFF + 32 * HSTR * 4;    // bf16 h shadow [32][HB]  16896 B
constexpr int SKVF_OFF = SHB_OFF + 32 * HB * 2;     // f32 kV C-frag [8192]    32768 B
constexpr int SKEY_OFF = SKVF_OFF + 8192 * 4;       // f32 keys[20][HSTR]      20800 B
constexpr int SX_OFF   = SKEY_OFF + KK * HSTR * 4;  // f32 x[256]
constexpr int SXAF_OFF = SX_OFF + DD * 4;           // bf16 x A-frag staging    8192 B
constexpr int SXW_OFF  = SXAF_OFF + 8 * 512 * 2;    // f32 xW[256]
constexpr int SG_OFF   = SXW_OFF + DD * 4;          // f32 g[32]
constexpr int SNRM_OFF = SG_OFF + 32 * 4;           // f32 norm[32]
constexpr int SMEM_BYTES = SNRM_OFF + 32 * 4;       // ~245 KB (< 320 KB / WGP)

__device__ __forceinline__ __bf16 f2bf(float x) { return (__bf16)x; }

__global__ __launch_bounds__(256, 1)
void entity_rnn_wmma_kernel(const float* __restrict__ enc,           // [B,S,D]
                            const unsigned char* __restrict__ mask,  // [B,S] bool
                            const float* __restrict__ keys,          // [B,K,D]
                            const float* __restrict__ U,             // [D,D]
                            const float* __restrict__ V,             // [D,D]
                            const float* __restrict__ W,             // [D,D]
                            float* __restrict__ out)                 // [B,K,D]
{
    extern __shared__ __align__(16) char smem_raw[];
    __bf16* sWf   = (__bf16*)(smem_raw + SWF_OFF);
    float*  sh    = (float*)(smem_raw + SH_OFF);
    __bf16* shb   = (__bf16*)(smem_raw + SHB_OFF);
    float*  skvf  = (float*)(smem_raw + SKVF_OFF);
    float*  skeys = (float*)(smem_raw + SKEY_OFF);
    float*  sx    = (float*)(smem_raw + SX_OFF);
    __bf16* sxAf  = (__bf16*)(smem_raw + SXAF_OFF);
    float*  sxw   = (float*)(smem_raw + SXW_OFF);
    float*  sg    = (float*)(smem_raw + SG_OFF);
    float*  snorm = (float*)(smem_raw + SNRM_OFF);

    const int tid    = threadIdx.x;        // 256 threads = 8 wave32
    const int lane   = tid & 31;
    const int wid    = tid >> 5;           // 0..7
    const int mi     = wid >> 2;           // M-tile of h@U: 0 or 1
    const int niBase = (wid & 3) * 4;      // 4 N-tiles per wave for h@U
    const int half   = lane >> 4;          // 0/1
    const int lr     = lane & 15;
    const int b      = blockIdx.x;

    // ---- one-time block setup ---------------------------------------------
    // W -> LDS pre-packed in WMMA B-fragment order.
    for (int idx = tid; idx < DD * DD; idx += 256) {
        const int i  = idx & 15;
        const int l  = (idx >> 4) & 31;
        const int fg = idx >> 9;           // kt*16 + nt
        const int kt = fg >> 4, nt = fg & 15;
        sWf[idx] = f2bf(W[(kt * 32 + (l >> 4) * 16 + i) * DD + nt * 16 + (l & 15)]);
    }
    for (int i = tid; i < 32 * HSTR; i += 256) sh[i] = 0.f;
    for (int i = tid; i < 8192; i += 256) skvf[i] = 0.f;
    for (int i = tid; i < 32 * HB; i += 256) shb[i] = (__bf16)0.f;
    for (int i = tid; i < 8 * 512; i += 256) sxAf[i] = (__bf16)0.f;  // zeros persist
    for (int k = 0; k < KK; ++k)
        skeys[k * HSTR + tid] = keys[((size_t)b * KK + k) * DD + tid];
    if (tid < 32) { sg[tid] = 0.f; snorm[tid] = 0.f; }
    __syncthreads();

    // kV[k][e] = sum_d keys[k][d]*V[d][e]; store into C-fragment layout:
    // element (r,c) -> skvf[(( (r>>4)*4 + (c>>6) )*4 + ((c>>4)&3))*32*8
    //                       + (((r>>3)&1)*16 + (c&15))*8 + (r&7)]
    {
        float acc[KK];
#pragma unroll
        for (int k = 0; k < KK; ++k) acc[k] = 0.f;
        for (int d = 0; d < DD; ++d) {
            const float vd = V[d * DD + tid];
#pragma unroll
            for (int k = 0; k < KK; ++k)
                acc[k] = fmaf(skeys[k * HSTR + d], vd, acc[k]);
        }
        const int c = tid;
#pragma unroll
        for (int k = 0; k < KK; ++k) {
            const int w8 = (k >> 4) * 4 + (c >> 6);
            const int jn = (c >> 4) & 3;
            const int ln = ((k >> 3) & 1) * 16 + (c & 15);
            skvf[((w8 * 4 + jn) * 32 + ln) * 8 + (k & 7)] = acc[k];
        }
    }

    // Register-resident U B-fragments: 8 K-tiles x 4 N-tiles per wave.
    v16bf uf[32];
#pragma unroll
    for (int kt = 0; kt < 8; ++kt) {
#pragma unroll
        for (int jn = 0; jn < 4; ++jn) {
            const int col = (niBase + jn) * 16 + lr;
            const int k0  = kt * 32 + half * 16;
            v16bf f;
#pragma unroll
            for (int i = 0; i < 16; ++i) f[i] = f2bf(U[(k0 + i) * DD + col]);
            uf[kt * 4 + jn] = f;
        }
    }
    __syncthreads();

    // ---- sequential scan over S -------------------------------------------
    for (int t = 0; t < SS; ++t) {
        if (!mask[(size_t)b * SS + t]) continue;   // h unchanged -> skip step

        const float* xg = enc + ((size_t)b * SS + t) * DD;
        // stage 1: stage x (f32 for gate; live bf16 slots of A-frag staging)
        sx[tid] = xg[tid];
        {   // only lane slots 0 and 16 of each fragment are live (row 0 of A)
            const int kt = tid >> 5, rem = tid & 31, hf = rem >> 4, i = rem & 15;
            sxAf[kt * 512 + hf * 256 + i] =
                f2bf(xg[kt * 32 + hf * 16 + i]);
        }
        if (tid < 32) snorm[tid] = 0.f;
        __syncthreads();

        // stage 2a: gate g[k] = sigmoid(sum_d x[d]*(h[k][d]+keys[k][d]))
        // lane covers d = lane*8 .. lane*8+7 (contiguous -> b128 loads)
#pragma unroll
        for (int kk = 0; kk < 3; ++kk) {
            const int k = wid + kk * 8;
            if (k < KK) {
                const int d0 = lane * 8;
                float s = 0.f;
#pragma unroll
                for (int i = 0; i < 8; ++i)
                    s = fmaf(sx[d0 + i],
                             sh[k * HSTR + d0 + i] + skeys[k * HSTR + d0 + i], s);
                s += __shfl_xor(s, 16); s += __shfl_xor(s, 8);
                s += __shfl_xor(s, 4);  s += __shfl_xor(s, 2);
                s += __shfl_xor(s, 1);
                if (lane == 0) sg[k] = 1.0f / (1.0f + __expf(-s));
            }
        }

        // stage 2b: x@W on WMMA; wave w owns N-tiles 2w, 2w+1
        {
            v8f xacc[2] = {};
#pragma unroll
            for (int kt = 0; kt < 8; ++kt) {
                const __bf16* pa = sxAf + kt * 512 + lane * 16;
                v16bf a;
#pragma unroll
                for (int i = 0; i < 16; ++i) a[i] = pa[i];   // 2x ds_load_b128
#pragma unroll
                for (int jn = 0; jn < 2; ++jn) {
                    const int nt = wid * 2 + jn;
                    const __bf16* pb = sWf + ((kt * 16 + nt) * 32 + lane) * 16;
                    v16bf bf;
#pragma unroll
                    for (int i = 0; i < 16; ++i) bf[i] = pb[i];
                    xacc[jn] = __builtin_amdgcn_wmma_f32_16x16x32_bf16(
                        false, a, false, bf, (short)0, xacc[jn], false, false);
                }
            }
#pragma unroll
            for (int jn = 0; jn < 2; ++jn)
                if (half == 0) sxw[(wid * 2 + jn) * 16 + lr] = xacc[jn][0];
        }

        // stage 2c: h@U via WMMA (A = bf16 shadow of h, B = register-resident U)
        v8f acc[4] = {};
        {
            const __bf16* pA = shb + (mi * 16 + lr) * HB;
#pragma unroll
            for (int kt = 0; kt < 8; ++kt) {
                const int cb = kt * 32 + half * 16;
                v16bf a;
#pragma unroll
                for (int i = 0; i < 8; ++i) {
                    a[i]     = pA[cb + i];
                    a[8 + i] = pA[cb + 8 + i];
                }
#pragma unroll
                for (int jn = 0; jn < 4; ++jn)
                    acc[jn] = __builtin_amdgcn_wmma_f32_16x16x32_bf16(
                        false, a, false, uf[kt * 4 + jn], (short)0, acc[jn],
                        false, false);
            }
        }
        __syncthreads();   // sg, sxw ready for all waves

        // stage 3: h_tilda = relu(hU + kV + xW); upd = h + g*h_tilda;
        // j-outer so each lane reduces its 4 tiles before the cross-lane pass.
        float upd[4][8];
        const int kvBase = wid * 4 * 32 * 8 + lane * 8;   // + jn*256 + j
#pragma unroll
        for (int j = 0; j < 8; ++j) {
            const int r = mi * 16 + half * 8 + j;
            float s = 0.f;
#pragma unroll
            for (int jn = 0; jn < 4; ++jn) {
                const int c = (niBase + jn) * 16 + lr;
                float ht = acc[jn][j] + skvf[kvBase + jn * 256 + j] + sxw[c];
                ht = fmaxf(ht, 0.f);
                float u = fmaf(sg[r], ht, sh[r * HSTR + c]);
                upd[jn][j] = u;
                s = fmaf(u, u, s);
            }
            s += __shfl_xor(s, 1); s += __shfl_xor(s, 2);
            s += __shfl_xor(s, 4); s += __shfl_xor(s, 8);
            if (lr == 0) atomicAdd(&snorm[r], s);
        }
        __syncthreads();   // norms complete

        // stage 4: normalized writeback (f32 truth + bf16 shadow); rows>=K stay 0
#pragma unroll
        for (int j = 0; j < 8; ++j) {
            const int r = mi * 16 + half * 8 + j;
            if (r < KK) {
                const float rn = rsqrtf(fmaxf(snorm[r], EPSN));
#pragma unroll
                for (int jn = 0; jn < 4; ++jn) {
                    const int c = (niBase + jn) * 16 + lr;
                    const float hn = upd[jn][j] * rn;
                    sh[r * HSTR + c] = hn;
                    shb[r * HB + c]  = f2bf(hn);
                }
            }
        }
        if (t + 1 < SS)    // warm caches for next step's x row
            __builtin_prefetch(&enc[((size_t)b * SS + (t + 1)) * DD + tid], 0, 1);
        __syncthreads();
    }

    __syncthreads();
    for (int k = 0; k < KK; ++k)
        out[((size_t)b * KK + k) * DD + tid] = sh[k * HSTR + tid];
}

extern "C" void kernel_launch(void* const* d_in, const int* in_sizes, int n_in,
                              void* d_out, int out_size, void* d_ws, size_t ws_size,
                              hipStream_t stream) {
    (void)in_sizes; (void)n_in; (void)out_size; (void)d_ws; (void)ws_size;
    const float*         enc  = (const float*)d_in[0];
    const unsigned char* mask = (const unsigned char*)d_in[1];  // jax bool = 1 byte
    const float*         keys = (const float*)d_in[2];
    const float*         U    = (const float*)d_in[3];
    const float*         V    = (const float*)d_in[4];
    const float*         W    = (const float*)d_in[5];
    float*               out  = (float*)d_out;

    (void)hipFuncSetAttribute((const void*)entity_rnn_wmma_kernel,
                              hipFuncAttributeMaxDynamicSharedMemorySize,
                              SMEM_BYTES);
    entity_rnn_wmma_kernel<<<dim3(BB), dim3(256), SMEM_BYTES, stream>>>(
        enc, mask, keys, U, V, W, out);
}
